// MeanField_38208029065274
// MI455X (gfx1250) — matrix operand order repaired and checked by reading, not compiled, not run on hardware
//
#include <hip/hip_runtime.h>
#include <stdint.h>

typedef __attribute__((ext_vector_type(16))) _Float16 v16h;
typedef __attribute__((ext_vector_type(8)))  float    v8f;

#define HH 512
#define WW 512
#define NC 21
#define TW 64
#define TH 8

__device__ __forceinline__ int uidx(int b, int c, int y, int x) {
    return ((b * NC + c) * HH + y) * WW + x;
}

// q = softmax(-u) over the class axis, per pixel.
__global__ __launch_bounds__(256)
void mf_init_softmax(const float* __restrict__ u, float* __restrict__ q)
{
    int t = blockIdx.x * blockDim.x + threadIdx.x;
    const int total = 2 * HH * WW;
    if (t >= total) return;
    int b   = t / (HH * WW);
    int pix = t % (HH * WW);
    int y = pix / WW, x = pix % WW;

    float v[NC];
    float m = -3.4e38f;
    #pragma unroll
    for (int c = 0; c < NC; ++c) {
        float val = -u[uidx(b, c, y, x)];
        v[c] = val;
        m = fmaxf(m, val);
    }
    float s = 0.f;
    #pragma unroll
    for (int c = 0; c < NC; ++c) { v[c] = __expf(v[c] - m); s += v[c]; }
    float inv = 1.f / s;
    #pragma unroll
    for (int c = 0; c < NC; ++c) q[uidx(b, c, y, x)] = v[c] * inv;
}

// One mean-field iteration, fully fused:
//  async-copy q tile (+halo) into LDS (GLOBAL_LOAD_ASYNC_TO_LDS_B32, ASYNCcnt)
//  -> per-lane q_tild (lands directly in WMMA B layout)
//  -> 2x v_wmma_f32_16x16x32_f16 (label-context mixing, 21 padded to 32)
//  -> -u - q_hat -> softmax over 21 classes (pair-lane shfl), or u + q_hat on final iter.
__global__ __launch_bounds__(256)
void mf_iter(const float* __restrict__ u, const float* __restrict__ ew,
             const float* __restrict__ L, const float* __restrict__ qin,
             float* __restrict__ qout, int final_iter)
{
    __shared__ float qs[NC][TH + 2][TW + 2];   // 21*10*66*4 = 55,440 B

    const int b   = blockIdx.z;
    const int x0  = blockIdx.x * TW;
    const int y0  = blockIdx.y * TH;
    const int tid = threadIdx.x;

    // ---- stage q tile with 1-pixel zero halo, all 21 classes ----
    // In-range elements: async global->LDS copy (no VGPR round trip).
    // Out-of-range halo (image border blocks only): plain zero store to LDS.
    const int PLANE = (TH + 2) * (TW + 2);
    const int TOT   = NC * PLANE;
    const float* qb = qin + (size_t)b * NC * HH * WW;
    for (int i = tid; i < TOT; i += 256) {
        int k  = i / PLANE;
        int r  = i % PLANE;
        int yy = r / (TW + 2);
        int xx = r % (TW + 2);
        int gy = y0 + yy - 1;
        int gx = x0 + xx - 1;
        float* lp = &qs[k][yy][xx];
        if (gy >= 0 && gy < HH && gx >= 0 && gx < WW) {
            unsigned lds_off = (unsigned)(uintptr_t)lp;       // low 32 bits = LDS offset
            unsigned goff    = (unsigned)(((k * HH + gy) * WW + gx) * 4);
            asm volatile("global_load_async_to_lds_b32 %0, %1, %2"
                         :: "v"(lds_off), "v"(goff), "s"(qb) : "memory");
        } else {
            *lp = 0.f;
        }
    }

    const int  lane = tid & 31;
    const int  wv   = tid >> 5;         // wave id 0..7 -> tile row
    const int  n    = lane & 15;        // pixel-in-group / matrix row
    const bool hi   = lane >= 16;

    // ---- A operands: label_context (21x21) padded to two 16x32 f16 tiles ----
    // 16-bit A layout (ISA 7.12.2): lane m=lane%16; VGPR v<4: K=(hi?8:0)+2v+w,
    // VGPR v>=4: K=16+(hi?8:0)+2(v-4)+w.  (Overlaps async-copy latency.)
    v16h a0, a1;
    #pragma unroll
    for (int h = 0; h < 16; ++h) {
        int v  = h >> 1, w = h & 1;
        int kk = (v < 4) ? ((hi ? 8 : 0) + 2 * v + w)
                         : (16 + (hi ? 8 : 0) + 2 * (v - 4) + w);
        float e0 = (kk < NC) ? L[n * NC + kk] : 0.f;              // rows 0..15
        int   r1 = 16 + n;                                        // rows 16..31
        float e1 = (kk < NC && r1 < NC) ? L[r1 * NC + kk] : 0.f;
        a0[h] = (_Float16)e0;
        a1[h] = (_Float16)e1;
    }

    asm volatile("s_wait_asynccnt 0x0" ::: "memory");  // own async LDS writes done
    __syncthreads();                                   // ...visible to whole WG

    const int gy    = y0 + wv;
    const int yy    = wv + 1;
    const int kbase = hi ? 16 : 0;
    const int mb    = hi ? 8 : 0;
    const int DYv[8] = { 0, 0, 1, -1, 1,  1, -1, -1 };
    const int DXv[8] = { 1, -1, 0, 0, 1, -1,  1, -1 };

    for (int s = 0; s < 4; ++s) {
        const int gx = x0 + s * 16 + n;
        const int xl = s * 16 + n + 1;

        float ew8[8];
        #pragma unroll
        for (int d = 0; d < 8; ++d)
            ew8[d] = ew[((b * 8 + d) * HH + gy) * WW + gx];

        // ---- B operand: q_tild for this pixel; lane holds classes mb..mb+7, 16+mb..16+mb+7.
        // 16-bit B layout (matches A): h<8 K=(hi?8:0)+h, h>=8 K=16+(hi?8:0)+(h-8), n = lane%16.
        v16h bm;
        #pragma unroll
        for (int h = 0; h < 16; ++h) {
            int k = mb + h + (h >= 8 ? 8 : 0);
            float acc = 0.f;
            if (k < NC) {
                #pragma unroll
                for (int d = 0; d < 8; ++d)
                    acc += ew8[d] * qs[k][yy + DYv[d]][xl + DXv[d]];
                acc *= 0.125f;
            }
            bm[h] = (_Float16)acc;
        }

        v8f c0 = {};
        v8f c1 = {};
        c0 = __builtin_amdgcn_wmma_f32_16x16x32_f16(false, a0, false, bm,
                                                    (short)0, c0, false, false);
        c1 = __builtin_amdgcn_wmma_f32_16x16x32_f16(false, a1, false, bm,
                                                    (short)0, c1, false, false);

        // C/D layout: c0[r] = D[mb+r, n], c1[r] = D[16+mb+r, n] (valid only lo lanes).
        // Branch-free epilogue: every lane owns classes mb..mb+7; classes 16..20
        // are loaded unconditionally (hi lanes hit the same cachelines as their lo
        // partners) and masked to -FLT_MAX via cndmask AFTER the subtract (exp -> 0).
        float vals[8], val2[5];
        #pragma unroll
        for (int r = 0; r < 8; ++r)
            vals[r] = -u[uidx(b, mb + r, gy, gx)] - c0[r];
        #pragma unroll
        for (int r = 0; r < 5; ++r) {
            float t = -u[uidx(b, 16 + r, gy, gx)] - c1[r];  // unconditional load
            val2[r] = hi ? -3.4e38f : t;                    // select, no branch
        }

        if (final_iter) {            // out = u + q_hat = -(-u - q_hat)
            #pragma unroll
            for (int r = 0; r < 8; ++r)
                qout[uidx(b, mb + r, gy, gx)] = -vals[r];
            if (!hi) {
                #pragma unroll
                for (int r = 0; r < 5; ++r)
                    qout[uidx(b, 16 + r, gy, gx)] = -val2[r];
            }
        } else {
            float m = -3.4e38f;
            #pragma unroll
            for (int r = 0; r < 8; ++r) m = fmaxf(m, vals[r]);
            #pragma unroll
            for (int r = 0; r < 5; ++r) m = fmaxf(m, val2[r]);
            m = fmaxf(m, __shfl_xor(m, 16, 32));        // pair lanes n <-> n+16
            float ssum = 0.f;
            #pragma unroll
            for (int r = 0; r < 8; ++r) { vals[r] = __expf(vals[r] - m); ssum += vals[r]; }
            #pragma unroll
            for (int r = 0; r < 5; ++r) { val2[r] = __expf(val2[r] - m); ssum += val2[r]; }
            ssum += __shfl_xor(ssum, 16, 32);
            float inv = 1.f / ssum;
            #pragma unroll
            for (int r = 0; r < 8; ++r)
                qout[uidx(b, mb + r, gy, gx)] = vals[r] * inv;
            if (!hi) {
                #pragma unroll
                for (int r = 0; r < 5; ++r)
                    qout[uidx(b, 16 + r, gy, gx)] = val2[r] * inv;
            }
        }
    }
}

extern "C" void kernel_launch(void* const* d_in, const int* in_sizes, int n_in,
                              void* d_out, int out_size, void* d_ws, size_t ws_size,
                              hipStream_t stream)
{
    const float* unary = (const float*)d_in[0];   // (2,1,21,512,512)
    const float* ew    = (const float*)d_in[1];   // (2,8,512,512)
    const float* L     = (const float*)d_in[2];   // (21,21)
    float* out = (float*)d_out;                   // (2,1,21,512,512)
    float* qA  = (float*)d_ws;                    // 2*21*512*512 f32 = 44 MB scratch

    // q0 = softmax(-u)
    {
        const int total = 2 * HH * WW;
        mf_init_softmax<<<(total + 255) / 256, 256, 0, stream>>>(unary, qA);
    }

    // 5 mean-field iterations, ping-pong qA <-> out; final lands in out = u + q_hat.
    dim3 grid(WW / TW, HH / TH, 2);
    for (int it = 0; it < 5; ++it) {
        const float* src = (it % 2 == 0) ? qA  : out;
        float*       dst = (it % 2 == 0) ? out : qA;
        mf_iter<<<grid, 256, 0, stream>>>(unary, ew, L, src, dst, (it == 4) ? 1 : 0);
    }
}
